// multi_task_baseline_share_till_logit_70351564308758
// MI455X (gfx1250) — compile-verified, matrix operand order checked by reading, not run on hardware
//
#include <hip/hip_runtime.h>

// ---------------------------------------------------------------------------
// Fused  hidden = relu(pooled @ W1 + b1)  (split-bf16 x3 WMMA, fp32 accum)
//        -> hidden tile kept in LDS -> per-channel logit GEMM -> out[m][o][c]
//
//  * prep kernels convert W1 (transposed, K padded to 320) and w to bf16 in
//    d_ws once; hot kernel B-staging is a raw bf16 copy.
//  * B-staging uses GLOBAL_LOAD_ASYNC_TO_LDS_B128 (ASYNCcnt) when available.
//  * 2-deep LDS double buffering: async B copies + A global loads for chunk
//    k+1 are issued before the WMMAs of chunk k. LDS ~249 KB (WGP has 320 KB).
// ---------------------------------------------------------------------------

#define EMB   300
#define EMBP  320          // padded K for W1^T (zeros for k >= 300)
#define HID   512
#define NCHAN 32
#define NOUT  16
#define MPERC 4096
#define BM    64           // hidden rows per workgroup (one channel per WG)
#define BK    32           // K chunk
#define AST   40           // LDS row stride (elems) for staging tiles
#define HST   520          // LDS row stride for hidden tile

typedef __attribute__((ext_vector_type(16))) __bf16 bf16x16;
typedef __attribute__((ext_vector_type(8)))  float  f32x8;
typedef __attribute__((ext_vector_type(4)))  int    v4i;

union FragB16 {
    bf16x16 v;
    uint4 q[2];
    unsigned short u[16];
};

__device__ __forceinline__ unsigned short bf16_rne(float f) {
    unsigned int x = __float_as_uint(f);
    x += 0x7FFFu + ((x >> 16) & 1u);
    return (unsigned short)(x >> 16);
}
__device__ __forceinline__ float bf16_as_f32(unsigned short h) {
    return __uint_as_float(((unsigned int)h) << 16);
}
__device__ __forceinline__ bf16x16 load_frag_lds(const unsigned short* p) {
    FragB16 f;
    f.q[0] = *(const uint4*)(p);
    f.q[1] = *(const uint4*)(p + 8);
    return f.v;
}

// ---- CDNA5 async global->LDS copy (ASYNCcnt) with portable fallback --------
#if defined(__AMDGCN__) && __has_builtin(__builtin_amdgcn_global_load_async_to_lds_b128)
#define HAVE_ASYNC_LDS 1
typedef __attribute__((address_space(1))) v4i* gbl_v4i_p;   // "__device__" AS
typedef __attribute__((address_space(3))) v4i* lds_v4i_p;   // "__shared__" AS
__device__ __forceinline__ void async_cp16(void* lds_dst, const void* gsrc) {
    __builtin_amdgcn_global_load_async_to_lds_b128(
        (gbl_v4i_p)(unsigned long long)gsrc,
        (lds_v4i_p)(unsigned int)(unsigned long long)lds_dst,
        0, 0);
}
#if __has_builtin(__builtin_amdgcn_s_wait_asynccnt)
__device__ __forceinline__ void wait_async0() { __builtin_amdgcn_s_wait_asynccnt(0); }
#else
__device__ __forceinline__ void wait_async0() {
    asm volatile("s_wait_asynccnt 0x0" ::: "memory");
}
#endif
#else
#define HAVE_ASYNC_LDS 0
__device__ __forceinline__ void async_cp16(void* lds_dst, const void* gsrc) {
    *(uint4*)lds_dst = *(const uint4*)gsrc;
}
__device__ __forceinline__ void wait_async0() {}
#endif

__device__ __forceinline__ void conv_store4(unsigned short* dhi, unsigned short* dlo,
                                            float4 a) {
    unsigned short h0 = bf16_rne(a.x), h1 = bf16_rne(a.y),
                   h2 = bf16_rne(a.z), h3 = bf16_rne(a.w);
    unsigned short l0 = bf16_rne(a.x - bf16_as_f32(h0));
    unsigned short l1 = bf16_rne(a.y - bf16_as_f32(h1));
    unsigned short l2 = bf16_rne(a.z - bf16_as_f32(h2));
    unsigned short l3 = bf16_rne(a.w - bf16_as_f32(h3));
    uint2 hp, lp;
    hp.x = (unsigned)h0 | ((unsigned)h1 << 16);
    hp.y = (unsigned)h2 | ((unsigned)h3 << 16);
    lp.x = (unsigned)l0 | ((unsigned)l1 << 16);
    lp.y = (unsigned)l2 | ((unsigned)l3 << 16);
    *(uint2*)dhi = hp;
    *(uint2*)dlo = lp;
}

// ---------------------------------------------------------------------------
// Prep kernels: one-time f32 -> split-bf16 conversion of the shared weights.
// ---------------------------------------------------------------------------
__global__ __launch_bounds__(256)
void prep_w1t_kernel(const float* __restrict__ W1,
                     unsigned short* __restrict__ w1t_hi,
                     unsigned short* __restrict__ w1t_lo) {
    const int gid = blockIdx.x * 256 + threadIdx.x;   // 0 .. 320*512-1
    const int k = gid / HID;                          // consecutive lanes -> same k,
    const int n = gid % HID;                          // consecutive n (coalesced read)
    float v = (k < EMB) ? W1[(size_t)k * HID + n] : 0.0f;
    unsigned short h = bf16_rne(v);
    unsigned short l = bf16_rne(v - bf16_as_f32(h));
    w1t_hi[(size_t)n * EMBP + k] = h;                 // transposed [n][k]
    w1t_lo[(size_t)n * EMBP + k] = l;
}

__global__ __launch_bounds__(256)
void prep_wc_kernel(const float* __restrict__ w,
                    unsigned short* __restrict__ wcb) {
    const int gid = blockIdx.x * 256 + threadIdx.x;   // 0 .. 32*16*512-1
    wcb[gid] = bf16_rne(w[gid]);
}

// ---------------------------------------------------------------------------
// Fused main kernel.
// ---------------------------------------------------------------------------
__global__ __launch_bounds__(256, 1)
void fused_mlp_grouped_kernel(const float* __restrict__ pooled,
                              const unsigned short* __restrict__ w1t_hi,
                              const unsigned short* __restrict__ w1t_lo,
                              const float* __restrict__ b1,
                              const unsigned short* __restrict__ wcb,
                              const float* __restrict__ bb,
                              float* __restrict__ out)
{
    __shared__ unsigned short AsHi[2][BM * AST];
    __shared__ unsigned short AsLo[2][BM * AST];
    __shared__ unsigned short BsHi[2][HID * AST];
    __shared__ unsigned short BsLo[2][HID * AST];
    __shared__ unsigned short Hs[BM * HST];
    __shared__ float Red[4 * 256];

    const int tid   = threadIdx.x;
    const int wave  = tid >> 5;
    const int lane  = tid & 31;
    const int lrow  = lane & 15;
    const int khalf = lane >> 4;
    const int rowBase = blockIdx.x * BM;

    const int wm = wave & 1;       // 2 (M) x 4 (N) waves; wave tile 32 x 128
    const int wn = wave >> 1;

    f32x8 acc[2][8];
#pragma unroll
    for (int i = 0; i < 2; ++i)
#pragma unroll
        for (int j = 0; j < 8; ++j)
#pragma unroll
            for (int r = 0; r < 8; ++r) acc[i][j][r] = 0.0f;

    const int kq  = tid & 7;       // A loader: float4 slot in 32-wide K row
    const int ar0 = tid >> 3;      // A loader: row 0..31 (+32)
    const int seg = tid & 3;       // B loader: 16B segment in 64B row chunk
    const int br0 = tid >> 2;      // B loader: row 0..63 (+64*p)

    float4 rA[2];

    // -------- prologue: stage chunk 0 into buffer 0 --------
#pragma unroll
    for (int p = 0; p < 8; ++p) {
        const int n = br0 + p * 64;
        async_cp16(&BsHi[0][n * AST + seg * 8], w1t_hi + (size_t)n * EMBP + seg * 8);
        async_cp16(&BsLo[0][n * AST + seg * 8], w1t_lo + (size_t)n * EMBP + seg * 8);
    }
#pragma unroll
    for (int p = 0; p < 2; ++p) {
        const int r = ar0 + p * 32;
        rA[p] = *(const float4*)(pooled + (size_t)(rowBase + r) * EMB + kq * 4);
    }
#pragma unroll
    for (int p = 0; p < 2; ++p) {
        const int r = ar0 + p * 32;
        conv_store4(&AsHi[0][r * AST + kq * 4], &AsLo[0][r * AST + kq * 4], rA[p]);
    }
    wait_async0();
    __syncthreads();

    // -------- GEMM1 main loop: 10 chunks of K=32, double-buffered ----------
    for (int kc = 0; kc < 10; ++kc) {
        const int cur = kc & 1;
        const int nxt = cur ^ 1;

        if (kc < 9) {
            const int kbn = (kc + 1) * BK;
            // async B copies for next chunk (bf16, pre-transposed, pre-padded)
#pragma unroll
            for (int p = 0; p < 8; ++p) {
                const int n = br0 + p * 64;
                async_cp16(&BsHi[nxt][n * AST + seg * 8],
                           w1t_hi + (size_t)n * EMBP + kbn + seg * 8);
                async_cp16(&BsLo[nxt][n * AST + seg * 8],
                           w1t_lo + (size_t)n * EMBP + kbn + seg * 8);
            }
            // A global loads for next chunk (fp32; EMB % 4 == 0 -> all-or-none)
            const int kg = kbn + kq * 4;
#pragma unroll
            for (int p = 0; p < 2; ++p) {
                const int r = ar0 + p * 32;
                rA[p] = make_float4(0.f, 0.f, 0.f, 0.f);
                if (kg < EMB)
                    rA[p] = *(const float4*)(pooled + (size_t)(rowBase + r) * EMB + kg);
            }
        }

        // compute current chunk: 16 tiles x (hi*hi + hi*lo + lo*hi)
        bf16x16 ah[2], al[2];
#pragma unroll
        for (int mt = 0; mt < 2; ++mt) {
            const int ao = (wm * 32 + mt * 16 + lrow) * AST + khalf * 16;
            ah[mt] = load_frag_lds(&AsHi[cur][ao]);
            al[mt] = load_frag_lds(&AsLo[cur][ao]);
        }
#pragma unroll
        for (int nt = 0; nt < 8; ++nt) {
            const int bo = (wn * 128 + nt * 16 + lrow) * AST + khalf * 16;
            bf16x16 bh = load_frag_lds(&BsHi[cur][bo]);
            bf16x16 bl = load_frag_lds(&BsLo[cur][bo]);
#pragma unroll
            for (int mt = 0; mt < 2; ++mt) {
                acc[mt][nt] = __builtin_amdgcn_wmma_f32_16x16x32_bf16(
                    false, ah[mt], false, bh, (short)0, acc[mt][nt], false, false);
                acc[mt][nt] = __builtin_amdgcn_wmma_f32_16x16x32_bf16(
                    false, ah[mt], false, bl, (short)0, acc[mt][nt], false, false);
                acc[mt][nt] = __builtin_amdgcn_wmma_f32_16x16x32_bf16(
                    false, al[mt], false, bh, (short)0, acc[mt][nt], false, false);
            }
        }

        if (kc < 9) {
#pragma unroll
            for (int p = 0; p < 2; ++p) {
                const int r = ar0 + p * 32;
                conv_store4(&AsHi[nxt][r * AST + kq * 4],
                            &AsLo[nxt][r * AST + kq * 4], rA[p]);
            }
        }
        wait_async0();
        __syncthreads();
    }

    // -------- epilogue: bias + relu -> hidden tile (bf16) in LDS -----------
#pragma unroll
    for (int nt = 0; nt < 8; ++nt) {
        const int n = wn * 128 + nt * 16 + lrow;
        const float bias = b1[n];
#pragma unroll
        for (int mt = 0; mt < 2; ++mt) {
#pragma unroll
            for (int r = 0; r < 8; ++r) {
                const int m = wm * 32 + mt * 16 + r + khalf * 8;
                float v = acc[mt][nt][r] + bias;
                v = fmaxf(v, 0.0f);
                Hs[m * HST + n] = bf16_rne(v);
            }
        }
    }
    __syncthreads();

    // -------- GEMM2: [64 x 512] @ w[c][16 x 512]^T -------------------------
    const int c     = blockIdx.x >> 6;          // 64 blocks per channel
    const int mbase = (blockIdx.x & 63) * BM;
    const int mt2   = wave & 3;                 // M tile (16 rows)
    const int kh    = wave >> 2;                // K half (256 each)

    f32x8 acc2;
#pragma unroll
    for (int r = 0; r < 8; ++r) acc2[r] = 0.0f;

    const unsigned short* wrow = wcb + ((size_t)c * NOUT + lrow) * HID;  // row o=lrow
#pragma unroll
    for (int kc2 = 0; kc2 < 8; ++kc2) {
        const int k0 = kh * 256 + kc2 * 32 + khalf * 16;
        bf16x16 a2 = load_frag_lds(&Hs[(mt2 * 16 + lrow) * HST + k0]);
        FragB16 bw;
        bw.q[0] = *(const uint4*)(wrow + k0);
        bw.q[1] = *(const uint4*)(wrow + k0 + 8);
        acc2 = __builtin_amdgcn_wmma_f32_16x16x32_bf16(
            false, a2, false, bw.v, (short)0, acc2, false, false);
    }

    if (kh == 1) {
#pragma unroll
        for (int r = 0; r < 8; ++r) Red[mt2 * 256 + r * 32 + lane] = acc2[r];
    }
    __syncthreads();
    if (kh == 0) {
        const int o = lane & 15;
        const float bias2 = bb[c * NOUT + o];
#pragma unroll
        for (int r = 0; r < 8; ++r) {
            const float v = acc2[r] + Red[mt2 * 256 + r * 32 + lane] + bias2;
            const int m = mbase + mt2 * 16 + r + khalf * 8;
            out[((size_t)m * NOUT + o) * NCHAN + c] = v;   // pred_logit[m][o][c]
        }
    }
}

extern "C" void kernel_launch(void* const* d_in, const int* in_sizes, int n_in,
                              void* d_out, int out_size, void* d_ws, size_t ws_size,
                              hipStream_t stream) {
    const float* pooled = (const float*)d_in[0];
    const float* W1     = (const float*)d_in[1];
    const float* b1     = (const float*)d_in[2];
    const float* w      = (const float*)d_in[3];
    const float* bb     = (const float*)d_in[4];
    float* out = (float*)d_out;
    (void)in_sizes; (void)n_in; (void)out_size; (void)ws_size;

    unsigned short* ws16   = (unsigned short*)d_ws;
    unsigned short* w1t_hi = ws16;                                   // 512*320
    unsigned short* w1t_lo = ws16 + (size_t)HID * EMBP;              // 512*320
    unsigned short* wcb    = ws16 + (size_t)2 * HID * EMBP;          // 32*16*512

    prep_w1t_kernel<<<(HID * EMBP) / 256, 256, 0, stream>>>(W1, w1t_hi, w1t_lo);
    prep_wc_kernel<<<(NCHAN * NOUT * HID) / 256, 256, 0, stream>>>(w, wcb);

    const int nblocks = (NCHAN * MPERC) / BM;   // 2048
    fused_mlp_grouped_kernel<<<nblocks, 256, 0, stream>>>(
        pooled, w1t_hi, w1t_lo, b1, wcb, bb, out);
}
